// TtLlamaSDPA_43834436223585
// MI455X (gfx1250) — matrix operand order repaired and checked by reading, not compiled, not run on hardware
//
#include <hip/hip_runtime.h>
#include <hip/hip_bf16.h>

typedef __attribute__((ext_vector_type(16))) __bf16 v16bf;
typedef __attribute__((ext_vector_type(8)))  float  v8f;
typedef int v4i_vs __attribute__((vector_size(16)));   // async builtin pointee type

__device__ __forceinline__ __bf16 f2bf(float f) { return (__bf16)f; }  // v_cvt_pk_bf16_f32

// ---------------------------------------------------------------------------
// 16-lane butterfly reductions via DPP16 (pure VALU).
// ---------------------------------------------------------------------------
#if __has_builtin(__builtin_amdgcn_update_dpp)
template <int CTRL>
__device__ __forceinline__ float dppf(float x) {
    return __builtin_bit_cast(float,
        __builtin_amdgcn_update_dpp(0, __builtin_bit_cast(int, x),
                                    CTRL, 0xf, 0xf, true));
}
__device__ __forceinline__ float row16_max(float x) {
    x = fmaxf(x, dppf<0xB1>(x));
    x = fmaxf(x, dppf<0x4E>(x));
    x = fmaxf(x, dppf<0x141>(x));
    x = fmaxf(x, dppf<0x140>(x));
    return x;
}
__device__ __forceinline__ float row16_sum(float x) {
    x += dppf<0xB1>(x);
    x += dppf<0x4E>(x);
    x += dppf<0x141>(x);
    x += dppf<0x140>(x);
    return x;
}
#else
__device__ __forceinline__ float row16_max(float x) {
    x = fmaxf(x, __shfl_xor(x, 1, 32));
    x = fmaxf(x, __shfl_xor(x, 2, 32));
    x = fmaxf(x, __shfl_xor(x, 4, 32));
    x = fmaxf(x, __shfl_xor(x, 8, 32));
    return x;
}
__device__ __forceinline__ float row16_sum(float x) {
    x += __shfl_xor(x, 1, 32);
    x += __shfl_xor(x, 2, 32);
    x += __shfl_xor(x, 4, 32);
    x += __shfl_xor(x, 8, 32);
    return x;
}
#endif

#if __has_builtin(__builtin_amdgcn_global_load_async_to_lds_b128) && \
    __has_builtin(__builtin_amdgcn_s_wait_asynccnt)
#define USE_ASYNC_LDS 1
#define ASYNC_B128(g, l) \
    __builtin_amdgcn_global_load_async_to_lds_b128( \
        (__attribute__((address_space(1))) v4i_vs*)(g), \
        (__attribute__((address_space(3))) v4i_vs*)(l), 0, 0)
#else
#define USE_ASYNC_LDS 0
#endif

// ---------------------------------------------------------------------------
// Kernel 1: flash-attention (GQA), double-buffered K/V chunks.
// grid = B*H*(Q/128), block = 256 (8 waves); wave = 16 q-rows x 128 d.
// ---------------------------------------------------------------------------
__global__ __launch_bounds__(256) void sdpa_wmma_kernel(
    const float* __restrict__ xq, const float* __restrict__ keys,
    const float* __restrict__ values, const float* __restrict__ mask,
    __bf16* __restrict__ attn_out)
{
    const int H_ = 64, HK_ = 8, Qn = 1024, Kn = 1024, Dn = 128, HID = 8192;

    const int blk  = blockIdx.x;
    const int qb   = blk & 7;
    const int h    = (blk >> 3) & 63;
    const int b    = blk >> 9;
    const int hk   = h >> 3;
    const int tid  = threadIdx.x;
    const int wave = tid >> 5;
    const int lane = tid & 31;
    const int l16  = lane & 15;
    const int hi   = (lane >> 4) & 1;

    __shared__ __bf16 ksm[2][32][136];
    __shared__ __bf16 vsm[2][32][136];
    __shared__ __bf16 psm[8][16][32];

    const int qbase = qb * 128 + wave * 16;

    const float* qptr = xq     + ((size_t)b * H_  + h ) * Qn * Dn;
    const float* kptr = keys   + ((size_t)b * HK_ + hk) * Kn * Dn;
    const float* vptr = values + ((size_t)b * HK_ + hk) * Kn * Dn;
    const float* mptr = mask   + ((size_t)b * Qn) * Kn;

    // Q resident as 4 bf16 A-fragments (16 rows x 32 d each)
    v16bf qfrag[4];
    {
        const int qrow = qbase + l16;
        #pragma unroll
        for (int c = 0; c < 4; ++c)
            #pragma unroll
            for (int e = 0; e < 16; ++e) {
                int kl = e + hi * 16;
                qfrag[c][e] = f2bf(qptr[(size_t)qrow * Dn + c * 32 + kl]);
            }
    }

    v8f o[8];
    v8f zero = {};
    #pragma unroll
    for (int dt = 0; dt < 8; ++dt) o[dt] = zero;
    float mrow[8], lrow[8];
    #pragma unroll
    for (int i = 0; i < 8; ++i) { mrow[i] = -1e30f; lrow[i] = 0.f; }

    const float scale = 0.08838834764831845f;  // 1/sqrt(128)

    const int srow  = tid >> 3;          // staging: 32 rows x 128 cols, 16/thread
    const int scolb = (tid & 7) << 4;

    auto stage = [&](int k0, int buf) {
        const float* kg = kptr + (size_t)(k0 + srow) * Dn + scolb;
        const float* vg = vptr + (size_t)(k0 + srow) * Dn + scolb;
        #pragma unroll
        for (int j = 0; j < 16; ++j) {
            ksm[buf][srow][scolb + j] = f2bf(kg[j]);
            vsm[buf][srow][scolb + j] = f2bf(vg[j]);
        }
    };

    stage(0, 0);

    for (int t = 0; t < Kn / 32; ++t) {
        const int cur = t & 1;
        const int k0  = t * 32;

        // mask loads: independent, issued early to overlap staging + WMMA
        const int kc0 = k0 + l16, kc1 = k0 + 16 + l16;
        float mk0[8], mk1[8];
        #pragma unroll
        for (int i = 0; i < 8; ++i) {
            int q = qbase + i + hi * 8;
            mk0[i] = mptr[(size_t)q * Kn + kc0];
            mk1[i] = mptr[(size_t)q * Kn + kc1];
        }

        __syncthreads();                       // chunk t ready; buf[cur^1] free
        if (t + 1 < Kn / 32) stage(k0 + 32, cur ^ 1);
        if (t + 2 < Kn / 32) {
            __builtin_prefetch(kptr + (size_t)(k0 + 64 + srow) * Dn, 0, 1);
            __builtin_prefetch(vptr + (size_t)(k0 + 64 + srow) * Dn, 0, 1);
        }

        // ---- S = Q * K^T ----
        v8f s0 = zero, s1 = zero;
        #pragma unroll
        for (int c = 0; c < 4; ++c) {
            v16bf b0, b1;
            #pragma unroll
            for (int e = 0; e < 16; ++e) {
                int dl = c * 32 + e + hi * 16;
                b0[e] = ksm[cur][l16][dl];
                b1[e] = ksm[cur][16 + l16][dl];
            }
            s0 = __builtin_amdgcn_wmma_f32_16x16x32_bf16(false, qfrag[c], false, b0,
                                                         (short)0, s0, false, false);
            s1 = __builtin_amdgcn_wmma_f32_16x16x32_bf16(false, qfrag[c], false, b1,
                                                         (short)0, s1, false, false);
        }

        // ---- scale + mask + online softmax ----
        float mnew[8], al[8];
        v8f p0, p1;
        #pragma unroll
        for (int i = 0; i < 8; ++i) {
            float v0 = s0[i] * scale + mk0[i];
            float v1 = s1[i] * scale + mk1[i];
            s0[i] = v0; s1[i] = v1;
            float r = row16_max(fmaxf(v0, v1));
            mnew[i] = fmaxf(mrow[i], r);
            al[i]   = __expf(mrow[i] - mnew[i]);
            mrow[i] = mnew[i];
        }
        #pragma unroll
        for (int i = 0; i < 8; ++i) {
            float e0 = __expf(s0[i] - mnew[i]);
            float e1 = __expf(s1[i] - mnew[i]);
            p0[i] = e0; p1[i] = e1;
            lrow[i] = lrow[i] * al[i] + row16_sum(e0 + e1);
        }
        #pragma unroll
        for (int dt = 0; dt < 8; ++dt)
            #pragma unroll
            for (int i = 0; i < 8; ++i) o[dt][i] *= al[i];

        // ---- P: C-layout -> A-layout via per-wave LDS bounce ----
        #pragma unroll
        for (int i = 0; i < 8; ++i) {
            int r = i + hi * 8;
            psm[wave][r][l16]      = f2bf(p0[i]);
            psm[wave][r][16 + l16] = f2bf(p1[i]);
        }
        v16bf pa;
        #pragma unroll
        for (int e = 0; e < 16; ++e) {
            int kl = e + hi * 16;
            pa[e] = psm[wave][l16][kl];
        }

        // ---- O += P * V ----
        #pragma unroll
        for (int dt = 0; dt < 8; ++dt) {
            v16bf bv;
            #pragma unroll
            for (int e = 0; e < 16; ++e)
                bv[e] = vsm[cur][e + hi * 16][dt * 16 + l16];
            o[dt] = __builtin_amdgcn_wmma_f32_16x16x32_bf16(false, pa, false, bv,
                                                            (short)0, o[dt], false, false);
        }
    }

    const size_t obase = (size_t)b * Qn * HID + (size_t)h * Dn;
    #pragma unroll
    for (int i = 0; i < 8; ++i) {
        int q = qbase + i + hi * 8;
        float inv = 1.0f / lrow[i];
        #pragma unroll
        for (int dt = 0; dt < 8; ++dt)
            attn_out[obase + (size_t)q * HID + dt * 16 + l16] = f2bf(o[dt][i] * inv);
    }
}

// ---------------------------------------------------------------------------
// Kernel: streaming fp32 -> bf16 convert (for wo; one pass, then L2-resident).
// ---------------------------------------------------------------------------
__global__ __launch_bounds__(256) void cvt_bf16_kernel(
    const float* __restrict__ src, __bf16* __restrict__ dst)
{
    size_t i = ((size_t)blockIdx.x * 256 + threadIdx.x) * 16;
    #pragma unroll
    for (int j = 0; j < 16; ++j) dst[i + j] = f2bf(src[i + j]);
}

// ---------------------------------------------------------------------------
// Kernel 2a: out = A(bf16 2048x8192) @ W(bf16 8192x8192), fp32 out.
// Block tile 256x128, wave tile 32x128, K-step 64: 32 WMMAs per barrier.
// Double-buffered LDS, all staging via GLOBAL_LOAD_ASYNC_TO_LDS_B128
// (ASYNCcnt), one barrier per K-step.  Both operands (134 MB + 33.5 MB) fit
// the 192 MB global L2 -> compute-bound after first touch.
// grid = (N/128, M/256) = (64, 8), block = 256.
// ---------------------------------------------------------------------------
__global__ __launch_bounds__(256) void proj_wmma_bf16_kernel(
    const __bf16* __restrict__ A, const __bf16* __restrict__ W,
    float* __restrict__ out)
{
    const int N = 8192, KD = 8192, KSTEP = 64, NT = KD / KSTEP;

    const int tid  = threadIdx.x;
    const int wave = tid >> 5;
    const int lane = tid & 31;
    const int l16  = lane & 15;
    const int hi   = (lane >> 4) & 1;

    __shared__ __bf16 As[2][256][72];   // 256 x 64 (+8 pad; 144B rows, 16B-aligned)
    __shared__ __bf16 Ws[2][64][136];   // 64 x 128 (+8 pad; 272B rows, 16B-aligned)

    const int m0 = blockIdx.y * 256;
    const int n0 = blockIdx.x * 128;

    const int wrow  = tid >> 2;         // W staging: 64 rows, 32 cols/thread
    const int wcolb = (tid & 3) << 5;

    auto stage = [&](int kk, int buf) {
        const __bf16* ag = A + (size_t)(m0 + tid) * KD + kk;
        const __bf16* wg = W + (size_t)(kk + wrow) * N + n0 + wcolb;
#if USE_ASYNC_LDS
        #pragma unroll
        for (int c = 0; c < 8; ++c)
            ASYNC_B128(ag + c * 8, &As[buf][tid][c * 8]);
        #pragma unroll
        for (int c = 0; c < 4; ++c)
            ASYNC_B128(wg + c * 8, &Ws[buf][wrow][wcolb + c * 8]);
#else
        #pragma unroll
        for (int j = 0; j < 64; ++j) As[buf][tid][j] = ag[j];
        #pragma unroll
        for (int j = 0; j < 32; ++j) Ws[buf][wrow][wcolb + j] = wg[j];
#endif
    };

    v8f acc[16];
    v8f zero = {};
    #pragma unroll
    for (int i = 0; i < 16; ++i) acc[i] = zero;

    stage(0, 0);

    for (int t = 0; t < NT; ++t) {
        const int cur = t & 1;
#if USE_ASYNC_LDS
        __builtin_amdgcn_s_wait_asynccnt(0);
#endif
        __syncthreads();                      // tile t visible; buf[cur^1] free
        if (t + 1 < NT) stage((t + 1) * KSTEP, cur ^ 1);
        if (t + 2 < NT) {
            __builtin_prefetch(A + (size_t)(m0 + tid) * KD + (t + 2) * KSTEP, 0, 1);
            __builtin_prefetch(W + (size_t)((t + 2) * KSTEP + wrow) * N + n0, 0, 1);
        }

        #pragma unroll
        for (int sub = 0; sub < 2; ++sub) {   // two 32-deep sub-chunks
            v16bf af0, af1;
            #pragma unroll
            for (int e = 0; e < 16; ++e) {
                int kl = sub * 32 + e + hi * 16;
                af0[e] = As[cur][wave * 32 + l16][kl];
                af1[e] = As[cur][wave * 32 + 16 + l16][kl];
            }
            #pragma unroll
            for (int nt = 0; nt < 8; ++nt) {
                v16bf bw;
                #pragma unroll
                for (int e = 0; e < 16; ++e)
                    bw[e] = Ws[cur][sub * 32 + e + hi * 16][nt * 16 + l16];
                acc[nt]     = __builtin_amdgcn_wmma_f32_16x16x32_bf16(
                                  false, af0, false, bw, (short)0, acc[nt], false, false);
                acc[8 + nt] = __builtin_amdgcn_wmma_f32_16x16x32_bf16(
                                  false, af1, false, bw, (short)0, acc[8 + nt], false, false);
            }
        }
    }

    #pragma unroll
    for (int half = 0; half < 2; ++half)
        #pragma unroll
        for (int i = 0; i < 8; ++i) {
            int m = m0 + wave * 32 + half * 16 + i + hi * 8;
            #pragma unroll
            for (int nt = 0; nt < 8; ++nt)
                out[(size_t)m * N + n0 + nt * 16 + l16] = acc[half * 8 + nt][i];
        }
}

// ---------------------------------------------------------------------------
// Kernel 2b (fallback, small ws): W converted fp32->bf16 on the fly.
// ---------------------------------------------------------------------------
__global__ __launch_bounds__(256) void proj_wmma_f32w_kernel(
    const __bf16* __restrict__ A, const float* __restrict__ W,
    float* __restrict__ out)
{
    const int N = 8192, KD = 8192;

    const int tid  = threadIdx.x;
    const int wave = tid >> 5;
    const int lane = tid & 31;
    const int l16  = lane & 15;
    const int hi   = (lane >> 4) & 1;

    __shared__ __bf16 As[256][40];
    __shared__ __bf16 Ws[32][136];

    const int m0 = blockIdx.y * 256;
    const int n0 = blockIdx.x * 128;

    v8f acc[16];
    v8f zero = {};
    #pragma unroll
    for (int i = 0; i < 16; ++i) acc[i] = zero;

    for (int kk = 0; kk < KD; kk += 32) {
        {
            const __bf16* ag = A + (size_t)(m0 + tid) * KD + kk;
#if USE_ASYNC_LDS
            #pragma unroll
            for (int c = 0; c < 4; ++c)
                ASYNC_B128(ag + c * 8, &As[tid][c * 8]);
#else
            #pragma unroll
            for (int j = 0; j < 32; ++j) As[tid][j] = ag[j];
#endif
        }
        {
            int wrow = tid >> 3, wcolb = (tid & 7) << 4;
            const float* wg = W + (size_t)(kk + wrow) * N + n0 + wcolb;
            #pragma unroll
            for (int j = 0; j < 16; ++j) Ws[wrow][wcolb + j] = f2bf(wg[j]);
        }
#if USE_ASYNC_LDS
        __builtin_amdgcn_s_wait_asynccnt(0);
#endif
        __syncthreads();

        v16bf af0, af1;
        #pragma unroll
        for (int e = 0; e < 16; ++e) {
            int kl = e + hi * 16;
            af0[e] = As[wave * 32 + l16][kl];
            af1[e] = As[wave * 32 + 16 + l16][kl];
        }
        #pragma unroll
        for (int nt = 0; nt < 8; ++nt) {
            v16bf bw;
            #pragma unroll
            for (int e = 0; e < 16; ++e)
                bw[e] = Ws[e + hi * 16][nt * 16 + l16];
            acc[nt]     = __builtin_amdgcn_wmma_f32_16x16x32_bf16(false, af0, false, bw,
                                                                  (short)0, acc[nt], false, false);
            acc[8 + nt] = __builtin_amdgcn_wmma_f32_16x16x32_bf16(false, af1, false, bw,
                                                                  (short)0, acc[8 + nt], false, false);
        }
        __syncthreads();
    }

    #pragma unroll
    for (int half = 0; half < 2; ++half)
        #pragma unroll
        for (int i = 0; i < 8; ++i) {
            int m = m0 + wave * 32 + half * 16 + i + hi * 8;
            #pragma unroll
            for (int nt = 0; nt < 8; ++nt)
                out[(size_t)m * N + n0 + nt * 16 + l16] = acc[half * 8 + nt][i];
        }
}

extern "C" void kernel_launch(void* const* d_in, const int* in_sizes, int n_in,
                              void* d_out, int out_size, void* d_ws, size_t ws_size,
                              hipStream_t stream) {
    const float* xq     = (const float*)d_in[0];
    const float* keys   = (const float*)d_in[1];
    const float* values = (const float*)d_in[2];
    const float* mask   = (const float*)d_in[3];
    const float* wo     = (const float*)d_in[4];
    float* out = (float*)d_out;

    const size_t attn_bytes = (size_t)2 * 1024 * 8192 * sizeof(__bf16);   // 33.5 MB
    const size_t wo_bytes   = (size_t)8192 * 8192 * sizeof(__bf16);       // 134 MB

    __bf16* attn = (__bf16*)d_ws;

    sdpa_wmma_kernel<<<dim3(1024), dim3(256), 0, stream>>>(xq, keys, values, mask, attn);

    if (ws_size >= attn_bytes + wo_bytes) {
        __bf16* wo_bf = (__bf16*)((char*)d_ws + attn_bytes);
        // 8192*8192 / (256*16) = 16384 blocks
        cvt_bf16_kernel<<<dim3(16384), dim3(256), 0, stream>>>(wo, wo_bf);
        proj_wmma_bf16_kernel<<<dim3(64, 8), dim3(256), 0, stream>>>(attn, wo_bf, out);
    } else {
        proj_wmma_f32w_kernel<<<dim3(64, 8), dim3(256), 0, stream>>>(attn, wo, out);
    }
}
